// MultiheadLatentAttention_86620900426243
// MI455X (gfx1250) — matrix operand order repaired and checked by reading, not compiled, not run on hardware
//
#include <hip/hip_runtime.h>

typedef __bf16 bf16_t;
typedef __attribute__((ext_vector_type(16))) __bf16 v16bf;
typedef __attribute__((ext_vector_type(8)))  __bf16 v8bf;
typedef __attribute__((ext_vector_type(8)))  float  v8f;

__device__ __forceinline__ unsigned short f32_to_bf16_bits(float f) {
    unsigned int u = __builtin_bit_cast(unsigned int, f);
    u += 0x7FFFu + ((u >> 16) & 1u);               // round-to-nearest-even
    return (unsigned short)(u >> 16);
}

__device__ __forceinline__ v8f vzero8() {
    v8f z;
    #pragma unroll
    for (int i = 0; i < 8; ++i) z[i] = 0.0f;
    return z;
}

// A-fragment (16x32 bf16, contiguous k order, same as B): lane m = lane&15, half = lane>>4,
// element e -> K = half*16 + e. Two contiguous 16B loads.
__device__ __forceinline__ v16bf pack_a(const bf16_t* row, int kk, int hl) {
    v8bf lo = *(const v8bf*)(row + kk + hl * 16);
    v8bf hi = *(const v8bf*)(row + kk + hl * 16 + 8);
    v16bf a;
    #pragma unroll
    for (int e = 0; e < 8; ++e) { a[e] = lo[e]; a[e + 8] = hi[e]; }
    return a;
}

// ---------------------------------------------------------------------------
// Generic batched GEMM:  C[z](M,N) = A[z](M,K) * B[z](N,K)^T, bf16 in, f32 acc.
// z = z1*Z0 + z0 with independent strides so (batch,head) products map cleanly.
// Workgroup = 4 waves (2x2); WG tile 64x128, wave tile 32x64 = 2x4 WMMA tiles
// (8 WMMAs per K-step for 6 fragment loads -> higher intensity for the
// compute-bound 2048^3 projections). All M multiples of 64, N multiples of 64,
// K multiples of 32 in every call below; guard is wave-granular.
// ---------------------------------------------------------------------------
template<bool OUT_BF16>
__global__ __launch_bounds__(128) void gemm_bt(
    const bf16_t* __restrict__ A, const bf16_t* __restrict__ B, void* __restrict__ Cv,
    int M, int N, int K, int lda, int ldb, int ldc,
    long sA0, long sA1, long sB0, long sB1, long sC0, long sC1, int Z0)
{
    int z  = blockIdx.z;
    int z0 = z % Z0, z1 = z / Z0;
    A += (long)z0 * sA0 + (long)z1 * sA1;
    B += (long)z0 * sB0 + (long)z1 * sB1;
    long cOff = (long)z0 * sC0 + (long)z1 * sC1;

    int lane = threadIdx.x & 31;
    int wave = threadIdx.x >> 5;
    int hl   = lane >> 4;
    int l16  = lane & 15;

    int bm = blockIdx.x * 64  + (wave >> 1) * 32;
    int bn = blockIdx.y * 128 + (wave & 1) * 64;
    if (bm >= M || bn >= N) return;

    v8f acc[2][4];
    #pragma unroll
    for (int i = 0; i < 2; ++i)
        #pragma unroll
        for (int j = 0; j < 4; ++j) acc[i][j] = vzero8();

    const bf16_t* Ar[2];
    const bf16_t* Br[4];
    #pragma unroll
    for (int i = 0; i < 2; ++i) Ar[i] = A + (long)(bm + i * 16 + l16) * lda;
    #pragma unroll
    for (int j = 0; j < 4; ++j) Br[j] = B + (long)(bn + j * 16 + l16) * ldb;

    for (int kk = 0; kk < K; kk += 32) {
        __builtin_prefetch(Ar[0] + kk + 128, 0, 0);   // -> global_prefetch_b8
        __builtin_prefetch(Br[0] + kk + 128, 0, 0);
        v16bf a[2], b[4];
        #pragma unroll
        for (int i = 0; i < 2; ++i) a[i] = pack_a(Ar[i], kk, hl);
        // B-fragment: element e -> K = half*16 + e : one contiguous 32B load.
        #pragma unroll
        for (int j = 0; j < 4; ++j) b[j] = *(const v16bf*)(Br[j] + kk + hl * 16);
        #pragma unroll
        for (int i = 0; i < 2; ++i)
            #pragma unroll
            for (int j = 0; j < 4; ++j)
                acc[i][j] = __builtin_amdgcn_wmma_f32_16x16x32_bf16(
                    false, a[i], false, b[j], (short)0, acc[i][j], false, false);
    }

    #pragma unroll
    for (int i = 0; i < 2; ++i)
        #pragma unroll
        for (int j = 0; j < 4; ++j) {
            int col = bn + j * 16 + l16;
            #pragma unroll
            for (int t = 0; t < 8; ++t) {
                int row = bm + i * 16 + t + hl * 8;   // C layout: vgpr t, half
                long idx = cOff + (long)row * ldc + col;
                if (OUT_BF16) ((unsigned short*)Cv)[idx] = f32_to_bf16_bits(acc[i][j][t]);
                else          ((float*)Cv)[idx] = acc[i][j][t];
            }
        }
}

// ---------------------------------------------------------------------------
// Elementwise f32 -> bf16
// ---------------------------------------------------------------------------
__global__ void cvt_f32_bf16(const float* __restrict__ src,
                             unsigned short* __restrict__ dst, long n) {
    long i = (long)blockIdx.x * blockDim.x + threadIdx.x;
    long stride = (long)gridDim.x * blockDim.x;
    for (; i < n; i += stride) dst[i] = f32_to_bf16_bits(src[i]);
}

// Transpose + convert: src[z][R][C] f32 -> dst[z][C][R] bf16 (LDS-tiled)
__global__ void cvt_tr_f32_bf16(const float* __restrict__ src,
                                unsigned short* __restrict__ dst, int R, int C) {
    __shared__ float tile[32][33];
    long zo = (long)blockIdx.z * R * C;
    int c0 = blockIdx.x * 32, r0 = blockIdx.y * 32;
    int tx = threadIdx.x, ty = threadIdx.y;   // 32 x 8
    for (int i = ty; i < 32; i += 8) {
        int r = r0 + i, c = c0 + tx;
        tile[i][tx] = (r < R && c < C) ? src[zo + (long)r * C + c] : 0.0f;
    }
    __syncthreads();
    for (int i = ty; i < 32; i += 8) {
        int c = c0 + i, r = r0 + tx;
        if (c < C && r < R) dst[zo + (long)c * R + r] = f32_to_bf16_bits(tile[tx][i]);
    }
}

// ---------------------------------------------------------------------------
// RoPE (r = 64) on rows of x[row][64]; pos = row % pos_mod. f32 in, bf16 out.
// ---------------------------------------------------------------------------
__global__ void rope_f32_bf16(const float* __restrict__ x,
                              unsigned short* __restrict__ out, int pos_mod) {
    long row = blockIdx.x;
    int d = threadIdx.x;                 // 0..63
    int pos = (int)(row % pos_mod);
    const float* xr = x + row * 64;
    int i = d & 31;
    float inv = __powf(10000.0f, -(2.0f * (float)i) / 64.0f);
    float ang = (float)pos * inv;
    float c = cosf(ang), s = sinf(ang);
    float v = (d < 32) ? (xr[d] * c - xr[d + 32] * s)
                       : (xr[d] * c + xr[d - 32] * s);
    out[row * 64 + d] = f32_to_bf16_bits(v);
}

// ---------------------------------------------------------------------------
// Flash-style MLA attention.
//   qr  : bf16 [H, B*S, 64]     (rope'd queries in retrieval space)
//   kr  : bf16 [B, H, L, 64]    (rope'd keys)
//   vt  : bf16 [B, d_c, L]      (kv_c transposed -> V as B[N,K])
//   ctx : bf16 [H, B*S, d_c]
// Workgroup = 4 waves, one 16-row query tile. Each wave owns 128 of d_c=512.
// L processed in chunks of 128 with online softmax; stats + P live in LDS.
// ---------------------------------------------------------------------------
__global__ __launch_bounds__(128) void mla_attn(
    const bf16_t* __restrict__ qr, const bf16_t* __restrict__ kr,
    const bf16_t* __restrict__ vt, unsigned short* __restrict__ ctx,
    int S, int L, int dc, int H)
{
    constexpr int CH = 128;
    __shared__ float sS[16 * CH];
    __shared__ __align__(32) unsigned short sP[16 * CH];
    __shared__ float sM[16], sLsum[16], sScale[16];

    int st = blockIdx.x, hh = blockIdx.y, b = blockIdx.z;
    int Bn = gridDim.z;
    int tid = threadIdx.x;
    int lane = tid & 31, wave = tid >> 5;
    int hl = lane >> 4, l16 = lane & 15;

    // Q fragments (K = r = 64 -> two 16x16x32 steps), loaded once per wave.
    long qrow = ((long)hh * Bn * S + (long)b * S + st * 16 + l16) * 64;
    v16bf aq[2];
    aq[0] = pack_a(qr + qrow, 0, hl);
    aq[1] = pack_a(qr + qrow, 32, hl);

    v8f acc[8];
    #pragma unroll
    for (int t = 0; t < 8; ++t) acc[t] = vzero8();
    if (tid < 16) { sM[tid] = -1e30f; sLsum[tid] = 0.0f; }
    __syncthreads();

    const long krbase = ((long)b * H + hh) * (long)L * 64;
    const bf16_t* vbase = vt + (long)b * dc * L;

    for (int lc = 0; lc < L; lc += CH) {
        // -- 1: scores for this chunk (each wave: 32 of the 128 L-columns) --
        #pragma unroll
        for (int tj = 0; tj < 2; ++tj) {
            int n = lc + wave * 32 + tj * 16 + l16;
            const bf16_t* kp = kr + krbase + (long)n * 64;
            v8f s = vzero8();
            #pragma unroll
            for (int kk = 0; kk < 64; kk += 32) {
                v16bf bf = *(const v16bf*)(kp + kk + hl * 16);
                s = __builtin_amdgcn_wmma_f32_16x16x32_bf16(
                        false, aq[kk >> 5], false, bf, (short)0, s, false, false);
            }
            int col = wave * 32 + tj * 16 + l16;
            #pragma unroll
            for (int t = 0; t < 8; ++t) sS[(t + hl * 8) * CH + col] = s[t];
        }
        __syncthreads();

        // -- 2: online-softmax row stats (16 threads, one per query row) --
        if (tid < 16) {
            const float* rs = sS + tid * CH;
            float cmax = -1e30f;
            for (int c = 0; c < CH; ++c) cmax = fmaxf(cmax, rs[c]);
            float mold = sM[tid];
            float mnew = fmaxf(mold, cmax);
            float scale = __expf(mold - mnew);
            float psum = 0.0f;
            for (int c = 0; c < CH; ++c) psum += __expf(rs[c] - mnew);
            sM[tid] = mnew;
            sScale[tid] = scale;
            sLsum[tid] = sLsum[tid] * scale + psum;
        }
        __syncthreads();

        // -- 3: write P (bf16) to LDS; rescale running accumulators --
        {
            int prow = tid >> 3;
            int pc0 = (tid & 7) * 16;
            float mnew = sM[prow];
            for (int c = 0; c < 16; ++c) {
                float p = __expf(sS[prow * CH + pc0 + c] - mnew);
                sP[prow * CH + pc0 + c] = f32_to_bf16_bits(p);
            }
        }
        float scl[8];
        #pragma unroll
        for (int jj = 0; jj < 8; ++jj) scl[jj] = sScale[jj + hl * 8];
        #pragma unroll
        for (int t = 0; t < 8; ++t)
            #pragma unroll
            for (int jj = 0; jj < 8; ++jj) acc[t][jj] *= scl[jj];
        __syncthreads();

        // -- 4: ctx += P(16xCH) @ V(CH x 128-slice) via WMMA --
        #pragma unroll
        for (int kk = 0; kk < CH; kk += 32) {
            v16bf ap = pack_a((const bf16_t*)sP + l16 * CH, kk, hl);
            #pragma unroll
            for (int nt = 0; nt < 8; ++nt) {
                int n = wave * 128 + nt * 16 + l16;
                v16bf bv = *(const v16bf*)(vbase + (long)n * L + lc + kk + hl * 16);
                acc[nt] = __builtin_amdgcn_wmma_f32_16x16x32_bf16(
                              false, ap, false, bv, (short)0, acc[nt], false, false);
            }
        }
        __syncthreads();
    }

    // Epilogue: divide by softmax denominator, store bf16 ctx [H, B*S, dc].
    float linv[8];
    #pragma unroll
    for (int jj = 0; jj < 8; ++jj) linv[jj] = 1.0f / sLsum[jj + hl * 8];
    long crow0 = (long)hh * Bn * S + (long)b * S + st * 16;
    #pragma unroll
    for (int nt = 0; nt < 8; ++nt) {
        int col = wave * 128 + nt * 16 + l16;
        #pragma unroll
        for (int jj = 0; jj < 8; ++jj) {
            int row = jj + hl * 8;
            ctx[(crow0 + row) * (long)dc + col] = f32_to_bf16_bits(acc[nt][jj] * linv[jj]);
        }
    }
}

// ---------------------------------------------------------------------------
extern "C" void kernel_launch(void* const* d_in, const int* in_sizes, int n_in,
                              void* d_out, int out_size, void* d_ws, size_t ws_size,
                              hipStream_t stream) {
    (void)n_in; (void)out_size; (void)ws_size;
    const float* hq_f    = (const float*)d_in[0];   // [B,S,2048]
    const float* kvc_f   = (const float*)d_in[1];   // [B,L,512]
    const float* Wq_f    = (const float*)d_in[2];   // [2048,2048]  (N,K)
    const float* wkcq_f  = (const float*)d_in[3];   // [16,128,1536]
    const float* Wqr_f   = (const float*)d_in[4];   // [16,1536,64]
    const float* Wkr_f   = (const float*)d_in[5];   // [16,512,64]
    const float* wkckv_f = (const float*)d_in[6];   // [16,128,512] (N,K per head)
    const float* Wo_f    = (const float*)d_in[7];   // [2048,2048]  (N,K)
    float* out = (float*)d_out;

    const int S = 1024, Lc = 1024, DQ = 2048, H = 16, KD = 128, DC = 512,
              DCQ = 1536, R = 64;
    const int B = in_sizes[0] / (S * DQ);
    const long BS = (long)B * S;

    char* wp = (char*)d_ws;
    auto alloc = [&](long bytes) -> void* {
        void* p = wp; wp += (bytes + 255) & ~255L; return p;
    };
    unsigned short* h_q    = (unsigned short*)alloc(BS * DQ * 2);
    unsigned short* wq     = (unsigned short*)alloc((long)DQ * DQ * 2);
    unsigned short* kvc    = (unsigned short*)alloc((long)B * Lc * DC * 2);
    unsigned short* kvcT   = (unsigned short*)alloc((long)B * Lc * DC * 2);
    unsigned short* wkcq   = (unsigned short*)alloc((long)H * KD * DCQ * 2);
    unsigned short* wqrT   = (unsigned short*)alloc((long)H * R * DCQ * 2);
    unsigned short* wkrT   = (unsigned short*)alloc((long)H * R * DC * 2);
    unsigned short* wkckv  = (unsigned short*)alloc((long)H * KD * DC * 2);
    unsigned short* wo     = (unsigned short*)alloc((long)DQ * DQ * 2);
    unsigned short* qhk    = (unsigned short*)alloc(BS * DQ * 2);
    unsigned short* wcombT = (unsigned short*)alloc((long)H * R * KD * 2);
    float*          qr32   = (float*)alloc((long)H * BS * R * 4);
    float*          kr32   = (float*)alloc((long)B * H * Lc * R * 4);
    unsigned short* qr16   = (unsigned short*)alloc((long)H * BS * R * 2);
    unsigned short* kr16   = (unsigned short*)alloc((long)B * H * Lc * R * 2);
    unsigned short* ctx    = (unsigned short*)alloc((long)H * BS * DC * 2);
    unsigned short* clat   = (unsigned short*)alloc(BS * DQ * 2);

    auto cvt = [&](const float* s, unsigned short* d, long n) {
        long nb = (n + 255) / 256;
        int blocks = nb > 16384 ? 16384 : (int)nb;
        cvt_f32_bf16<<<blocks, 256, 0, stream>>>(s, d, n);
    };
    auto g16 = [&](const unsigned short* A, const unsigned short* Bm, unsigned short* C,
                   int M, int N, int K, int lda, int ldb, int ldc,
                   long sA0, long sA1, long sB0, long sB1, long sC0, long sC1,
                   int Z0, int Z) {
        dim3 g((M + 63) / 64, (N + 127) / 128, Z);
        gemm_bt<true><<<g, 128, 0, stream>>>((const bf16_t*)A, (const bf16_t*)Bm,
            (void*)C, M, N, K, lda, ldb, ldc, sA0, sA1, sB0, sB1, sC0, sC1, Z0);
    };
    auto g32 = [&](const unsigned short* A, const unsigned short* Bm, float* C,
                   int M, int N, int K, int lda, int ldb, int ldc,
                   long sA0, long sA1, long sB0, long sB1, long sC0, long sC1,
                   int Z0, int Z) {
        dim3 g((M + 63) / 64, (N + 127) / 128, Z);
        gemm_bt<false><<<g, 128, 0, stream>>>((const bf16_t*)A, (const bf16_t*)Bm,
            (void*)C, M, N, K, lda, ldb, ldc, sA0, sA1, sB0, sB1, sC0, sC1, Z0);
    };

    // 1) Convert to bf16 (straight) ...
    cvt(hq_f,    h_q,   BS * DQ);
    cvt(Wq_f,    wq,    (long)DQ * DQ);
    cvt(kvc_f,   kvc,   (long)B * Lc * DC);
    cvt(wkcq_f,  wkcq,  (long)H * KD * DCQ);
    cvt(wkckv_f, wkckv, (long)H * KD * DC);
    cvt(Wo_f,    wo,    (long)DQ * DQ);
    // ... and transposed where [K,N] must become [N,K]:
    dim3 tb(32, 8);
    cvt_tr_f32_bf16<<<dim3(DC / 32, Lc / 32, B), tb, 0, stream>>>(kvc_f, kvcT, Lc, DC);
    cvt_tr_f32_bf16<<<dim3((R + 31) / 32, DCQ / 32, H), tb, 0, stream>>>(Wqr_f, wqrT, DCQ, R);
    cvt_tr_f32_bf16<<<dim3((R + 31) / 32, DC / 32, H), tb, 0, stream>>>(Wkr_f, wkrT, DC, R);

    // 2) q_hk = hidden_q @ Wq^T : [B*S, 2048]
    g16(h_q, wq, qhk, (int)BS, DQ, DQ, DQ, DQ, DQ, 0, 0, 0, 0, 0, 0, 1, 1);

    // 3) Weight absorption: WcombT[h] = W_qr[h]^T @ w_kc_q[h]^T : [R, KD]
    g16(wqrT, wkcq, wcombT, R, KD, DCQ, DCQ, DCQ, KD,
        (long)R * DCQ, 0, (long)KD * DCQ, 0, (long)R * KD, 0, H, H);

    // 4) q_r[h] = q_hk[:, h-slice] @ WcombT[h]^T : [H, B*S, R] (f32 for RoPE)
    g32(qhk, wcombT, qr32, (int)BS, R, KD, DQ, KD, R,
        (long)KD, 0, (long)R * KD, 0, BS * R, 0, H, H);

    // 5) k_r[b,h] = kv_c[b] @ W_kr[h] : [B, H, L, R] (f32 for RoPE)
    g32(kvc, wkrT, kr32, Lc, R, DC, DC, DC, R,
        0, (long)Lc * DC, (long)R * DC, 0, (long)Lc * R, (long)H * Lc * R, H, H * B);

    // 6) RoPE (f32 in, bf16 out)
    rope_f32_bf16<<<(int)(H * BS), 64, 0, stream>>>(qr32, qr16, S);
    rope_f32_bf16<<<(int)((long)B * H * Lc), 64, 0, stream>>>(kr32, kr16, Lc);

    // 7) Flash attention -> ctx bf16 [H, B*S, DC]
    mla_attn<<<dim3(S / 16, H, B), 128, 0, stream>>>(
        (const bf16_t*)qr16, (const bf16_t*)kr16, (const bf16_t*)kvcT, ctx,
        S, Lc, DC, H);

    // 8) ctx_lat[:, h-slice] = ctx[h] @ w_kc_kv[h]^T : [B*S, 2048]
    g16(ctx, wkckv, clat, (int)BS, KD, DC, DC, DC, DQ,
        BS * DC, 0, (long)KD * DC, 0, (long)KD, 0, H, H);

    // 9) out = ctx_lat @ Wo^T : [B*S, 2048] f32
    g32(clat, wo, out, (int)BS, DQ, DQ, DQ, DQ, DQ, 0, 0, 0, 0, 0, 0, 1, 1);
}
